// TransformerLayer_9208409882638
// MI455X (gfx1250) — matrix operand order repaired and checked by reading, not compiled, not run on hardware
//
#include <hip/hip_runtime.h>
#include <math.h>

// ---------------- problem constants ----------------
#define BB   2
#define TT   2048
#define DD   1024
#define HH   16
#define HDD  64
#define MM   (BB * TT)     // 4096 rows of the activation matrix
#define D4   (4 * DD)      // 4096

typedef __attribute__((ext_vector_type(16))) __bf16 v16bf;
typedef __attribute__((ext_vector_type(8)))  float  v8f;

union BFrag {
    v16bf v;
    unsigned int u[8];
};

static __device__ __forceinline__ v8f v8f_zero() {
    v8f z;
#pragma unroll
    for (int i = 0; i < 8; ++i) z[i] = 0.0f;
    return z;
}

// fp32 -> bf16 (round-to-nearest-even), stored as ushort
static __device__ __forceinline__ unsigned short f2bf(float f) {
    unsigned int u = __builtin_bit_cast(unsigned int, f);
    u += 0x7FFFu + ((u >> 16) & 1u);
    return (unsigned short)(u >> 16);
}

static __device__ __forceinline__ float gelu_tanh(float x) {
    const float c = 0.7978845608028654f;  // sqrt(2/pi)
    float x3 = x * x * x;
    return 0.5f * x * (1.0f + tanhf(c * (x + 0.044715f * x3)));
}

// reductions within a 16-lane half-wave (wave32; xor masks 1..8 stay in-half)
static __device__ __forceinline__ float rmax16(float v) {
#pragma unroll
    for (int o = 1; o < 16; o <<= 1) v = fmaxf(v, __shfl_xor(v, o, 32));
    return v;
}
static __device__ __forceinline__ float rsum16(float v) {
#pragma unroll
    for (int o = 1; o < 16; o <<= 1) v += __shfl_xor(v, o, 32);
    return v;
}

// ---------------- fp32 -> bf16 conversion ----------------
__global__ __launch_bounds__(256) void cvt_bf16_kernel(const float* __restrict__ in,
                                                       unsigned short* __restrict__ out,
                                                       int n) {
    int idx = blockIdx.x * blockDim.x + threadIdx.x;
    int stride = gridDim.x * blockDim.x;
    for (; idx < n; idx += stride) out[idx] = f2bf(in[idx]);
}

// ---------------- LayerNorm (row per block) -> bf16 ----------------
__global__ __launch_bounds__(256) void ln_bf16_kernel(const float* __restrict__ x,
                                                      const float* __restrict__ w,
                                                      const float* __restrict__ b,
                                                      unsigned short* __restrict__ out) {
    __shared__ float red[256];
    const int row = blockIdx.x;
    const int tid = threadIdx.x;
    float v[4];
    float s = 0.0f;
#pragma unroll
    for (int k = 0; k < 4; ++k) {
        int c = k * 256 + tid;
        v[k] = x[(size_t)row * DD + c];
        s += v[k];
    }
    red[tid] = s;
    __syncthreads();
    for (int st = 128; st > 0; st >>= 1) {
        if (tid < st) red[tid] += red[tid + st];
        __syncthreads();
    }
    float mean = red[0] * (1.0f / DD);
    __syncthreads();
    float sq = 0.0f;
#pragma unroll
    for (int k = 0; k < 4; ++k) { float d = v[k] - mean; sq += d * d; }
    red[tid] = sq;
    __syncthreads();
    for (int st = 128; st > 0; st >>= 1) {
        if (tid < st) red[tid] += red[tid + st];
        __syncthreads();
    }
    float rs = rsqrtf(red[0] * (1.0f / DD) + 1e-5f);
#pragma unroll
    for (int k = 0; k < 4; ++k) {
        int c = k * 256 + tid;
        out[(size_t)row * DD + c] = f2bf((v[k] - mean) * rs * w[c] + b[c]);
    }
}

// ---------------- fused residual add + LayerNorm -> bf16 (+x2 fp32) -------
__global__ __launch_bounds__(256) void resid_ln_bf16_kernel(const float* __restrict__ x,
                                                            const float* __restrict__ att,
                                                            const float* __restrict__ w,
                                                            const float* __restrict__ b,
                                                            float* __restrict__ x2,
                                                            unsigned short* __restrict__ out) {
    __shared__ float red[256];
    const int row = blockIdx.x;
    const int tid = threadIdx.x;
    float v[4];
    float s = 0.0f;
#pragma unroll
    for (int k = 0; k < 4; ++k) {
        int c = k * 256 + tid;
        size_t idx = (size_t)row * DD + c;
        v[k] = x[idx] + att[idx];
        x2[idx] = v[k];
        s += v[k];
    }
    red[tid] = s;
    __syncthreads();
    for (int st = 128; st > 0; st >>= 1) {
        if (tid < st) red[tid] += red[tid + st];
        __syncthreads();
    }
    float mean = red[0] * (1.0f / DD);
    __syncthreads();
    float sq = 0.0f;
#pragma unroll
    for (int k = 0; k < 4; ++k) { float d = v[k] - mean; sq += d * d; }
    red[tid] = sq;
    __syncthreads();
    for (int st = 128; st > 0; st >>= 1) {
        if (tid < st) red[tid] += red[tid + st];
        __syncthreads();
    }
    float rs = rsqrtf(red[0] * (1.0f / DD) + 1e-5f);
#pragma unroll
    for (int k = 0; k < 4; ++k) {
        int c = k * 256 + tid;
        out[(size_t)row * DD + c] = f2bf((v[k] - mean) * rs * w[c] + b[c]);
    }
}

// ---------------- WMMA NT GEMM:  C[M,N] = A[M,K] * W[N,K]^T  ----------------
// EPI 0: store bf16 as Q/K layout [B,H,T,HD]     (N must be 1024)
// EPI 1: store bf16 as V^T layout [B,H,HD,T]
// EPI 2: +bias, GELU, store bf16 row-major [M,N]
// EPI 3: +bias, +resid[M,N], store fp32 row-major [M,N]
template <int EPI>
__global__ __launch_bounds__(256) void gemm_nt_bf16(const unsigned short* __restrict__ A,
                                                    const unsigned short* __restrict__ W,
                                                    void* __restrict__ out,
                                                    int N, int K,
                                                    const float* __restrict__ bias,
                                                    const float* __restrict__ resid) {
    const int lane = threadIdx.x & 31;
    const int wid  = threadIdx.x >> 5;
    const int lh   = lane & 15;
    const bool hi  = lane >= 16;

    const int mbase = blockIdx.y * 128 + (wid & 3) * 32;
    const int nbase = blockIdx.x * 64 + (wid >> 2) * 32;

    int kaOff[8], kbOff[8];
#pragma unroll
    for (int j = 0; j < 8; ++j) {
        kaOff[j] = ((j < 4) ? 2 * j : 16 + 2 * (j - 4)) + (hi ? 8 : 0);
        kbOff[j] = ((j < 4) ? 2 * j : 16 + 2 * (j - 4)) + (hi ? 8 : 0);
    }

    v8f c[2][2];
#pragma unroll
    for (int mi = 0; mi < 2; ++mi)
#pragma unroll
        for (int ni = 0; ni < 2; ++ni) c[mi][ni] = v8f_zero();

    const unsigned short* Arow[2];
    const unsigned short* Wrow[2];
#pragma unroll
    for (int mi = 0; mi < 2; ++mi) Arow[mi] = A + (size_t)(mbase + 16 * mi + lh) * K;
#pragma unroll
    for (int ni = 0; ni < 2; ++ni) Wrow[ni] = W + (size_t)(nbase + 16 * ni + lh) * K;

    for (int kt = 0; kt < K; kt += 32) {
        BFrag a[2], bfr[2];
#pragma unroll
        for (int mi = 0; mi < 2; ++mi)
#pragma unroll
            for (int j = 0; j < 8; ++j)
                a[mi].u[j] = *(const unsigned int*)(Arow[mi] + kt + kaOff[j]);
#pragma unroll
        for (int ni = 0; ni < 2; ++ni)
#pragma unroll
            for (int j = 0; j < 8; ++j)
                bfr[ni].u[j] = *(const unsigned int*)(Wrow[ni] + kt + kbOff[j]);
#pragma unroll
        for (int mi = 0; mi < 2; ++mi)
#pragma unroll
            for (int ni = 0; ni < 2; ++ni)
                c[mi][ni] = __builtin_amdgcn_wmma_f32_16x16x32_bf16(
                    false, a[mi].v, false, bfr[ni].v, (short)0, c[mi][ni], false, false);
    }

    // epilogue; C layout: vgpr i / lanes0-15 -> row i, lanes16-31 -> row i+8; N = lh
#pragma unroll
    for (int mi = 0; mi < 2; ++mi) {
#pragma unroll
        for (int ni = 0; ni < 2; ++ni) {
#pragma unroll
            for (int i = 0; i < 8; ++i) {
                const int row = mbase + 16 * mi + i + (hi ? 8 : 0);
                const int col = nbase + 16 * ni + lh;
                float val = c[mi][ni][i];
                if constexpr (EPI == 0) {
                    int bi = row >> 11, t = row & (TT - 1);
                    int h = col >> 6, d = col & (HDD - 1);
                    ((unsigned short*)out)[(((size_t)(bi * HH + h) * TT) + t) * HDD + d] = f2bf(val);
                } else if constexpr (EPI == 1) {
                    int bi = row >> 11, t = row & (TT - 1);
                    int h = col >> 6, d = col & (HDD - 1);
                    ((unsigned short*)out)[(((size_t)(bi * HH + h) * HDD) + d) * TT + t] = f2bf(val);
                } else if constexpr (EPI == 2) {
                    val = gelu_tanh(val + bias[col]);
                    ((unsigned short*)out)[(size_t)row * N + col] = f2bf(val);
                } else {
                    val += bias[col] + resid[(size_t)row * N + col];
                    ((float*)out)[(size_t)row * N + col] = val;
                }
            }
        }
    }
}

// ---------------- flash attention with relative-position (skew) ----------------
// one wave32 per (b, h, 16-row q block); all-WMMA core, softmax state in registers
__global__ __launch_bounds__(32) void attn_kernel(const unsigned short* __restrict__ Qp,
                                                  const unsigned short* __restrict__ Kp,
                                                  const unsigned short* __restrict__ Vtp,
                                                  const unsigned short* __restrict__ Ep,
                                                  float* __restrict__ out) {
    __shared__ __align__(16) float Rlds[16 * 48];           // banded Q.Er^T tile
    __shared__ __align__(16) unsigned short Plds[16 * 32];  // P re-layout C->A frag

    const int blk  = blockIdx.x;
    const int qblk = blk & (TT / 16 - 1);
    const int h    = (blk >> 7) & (HH - 1);
    const int b    = blk >> 11;
    const int q0   = qblk * 16;

    const int lane = threadIdx.x & 31;
    const int lh   = lane & 15;
    const bool hi  = lane >= 16;

    int kaOff[8], kbOff[8];
#pragma unroll
    for (int j = 0; j < 8; ++j) {
        kaOff[j] = ((j < 4) ? 2 * j : 16 + 2 * (j - 4)) + (hi ? 8 : 0);
        kbOff[j] = ((j < 4) ? 2 * j : 16 + 2 * (j - 4)) + (hi ? 8 : 0);
    }

    // Q A-fragments (16 rows x 64 hd = 2 x (16x32)) straight from global bf16
    BFrag aQ[2];
    {
        const unsigned short* qrow = Qp + ((size_t)((b * HH + h) * TT) + q0 + lh) * HDD;
#pragma unroll
        for (int c2 = 0; c2 < 2; ++c2)
#pragma unroll
            for (int j = 0; j < 8; ++j)
                aQ[c2].u[j] = *(const unsigned int*)(qrow + 32 * c2 + kaOff[j]);
    }

    v8f o[4];
#pragma unroll
    for (int g = 0; g < 4; ++g) o[g] = v8f_zero();
    float mrow[8], lrow[8];
#pragma unroll
    for (int i = 0; i < 8; ++i) { mrow[i] = -1e30f; lrow[i] = 0.0f; }

    for (int k0 = 0; k0 <= q0 + 15; k0 += 32) {
        // ---- S = Q K^T  (2 frags x 2 K-chunks = 4 WMMA) ----
        v8f s[2];
#pragma unroll
        for (int f = 0; f < 2; ++f) {
            const unsigned short* krow =
                Kp + ((size_t)((b * HH + h) * TT) + k0 + 16 * f + lh) * HDD;
            v8f acc = v8f_zero();
#pragma unroll
            for (int c2 = 0; c2 < 2; ++c2) {
                BFrag bb;
#pragma unroll
                for (int j = 0; j < 8; ++j)
                    bb.u[j] = *(const unsigned int*)(krow + 32 * c2 + kbOff[j]);
                acc = __builtin_amdgcn_wmma_f32_16x16x32_bf16(
                    false, aQ[c2].v, false, bb.v, (short)0, acc, false, false);
            }
            s[f] = acc;
        }

        // ---- banded R = Q Er^T: srel[q,k] = Q[q].Er[T-1-q+k]  (6 WMMA) ----
        const int j0 = (TT - 16 - q0) + k0;
#pragma unroll
        for (int g = 0; g < 3; ++g) {
            int jc = j0 + 16 * g + lh;
            if (jc > TT - 1) jc = TT - 1;  // masked region only
            const unsigned short* erow = Ep + ((size_t)h * TT + jc) * HDD;
            v8f acc = v8f_zero();
#pragma unroll
            for (int c2 = 0; c2 < 2; ++c2) {
                BFrag bb;
#pragma unroll
                for (int j = 0; j < 8; ++j)
                    bb.u[j] = *(const unsigned int*)(erow + 32 * c2 + kbOff[j]);
                acc = __builtin_amdgcn_wmma_f32_16x16x32_bf16(
                    false, aQ[c2].v, false, bb.v, (short)0, acc, false, false);
            }
#pragma unroll
            for (int i = 0; i < 8; ++i) {
                int rr = i + (hi ? 8 : 0);
                Rlds[rr * 48 + 16 * g + lh] = acc[i];
            }
        }
        asm volatile("s_wait_dscnt 0x0" ::: "memory");

        // ---- diagonal gather + causal mask + online softmax ----
#pragma unroll
        for (int i = 0; i < 8; ++i) {
            const int rr = i + (hi ? 8 : 0);
            const int qrow = q0 + rr;
            float t0 = (s[0][i] + Rlds[rr * 48 + (lh + 15 - rr)]) * 0.125f;
            float t1 = (s[1][i] + Rlds[rr * 48 + (16 + lh + 15 - rr)]) * 0.125f;
            float v0 = (k0 + lh <= qrow) ? t0 : -1e30f;
            float v1 = (k0 + 16 + lh <= qrow) ? t1 : -1e30f;

            float mnew  = fmaxf(mrow[i], rmax16(fmaxf(v0, v1)));
            float scale = __expf(mrow[i] - mnew);
            float p0 = __expf(v0 - mnew);
            float p1 = __expf(v1 - mnew);
            lrow[i] = lrow[i] * scale + rsum16(p0 + p1);
            mrow[i] = mnew;
#pragma unroll
            for (int g = 0; g < 4; ++g) o[g][i] *= scale;
            Plds[rr * 32 + lh]      = f2bf(p0);
            Plds[rr * 32 + 16 + lh] = f2bf(p1);
        }
        asm volatile("s_wait_dscnt 0x0" ::: "memory");

        // ---- O += P V   (V stored transposed [B,H,HD,T]; 4 WMMA) ----
        BFrag aP;
#pragma unroll
        for (int j = 0; j < 8; ++j)
            aP.u[j] = *(const unsigned int*)&Plds[lh * 32 + kaOff[j]];
#pragma unroll
        for (int g = 0; g < 4; ++g) {
            const unsigned short* vrow =
                Vtp + ((size_t)((b * HH + h) * HDD) + 16 * g + lh) * TT;
            BFrag bb;
#pragma unroll
            for (int j = 0; j < 8; ++j)
                bb.u[j] = *(const unsigned int*)(vrow + k0 + kbOff[j]);
            o[g] = __builtin_amdgcn_wmma_f32_16x16x32_bf16(
                false, aP.v, false, bb.v, (short)0, o[g], false, false);
        }
    }

    // ---- normalize + write back to [B,T,D] fp32 ----
#pragma unroll
    for (int i = 0; i < 8; ++i) {
        const int rr = i + (hi ? 8 : 0);
        const int qrow = q0 + rr;
        float inv = 1.0f / lrow[i];
#pragma unroll
        for (int g = 0; g < 4; ++g)
            out[((size_t)(b * TT + qrow)) * DD + h * HDD + 16 * g + lh] = o[g][i] * inv;
    }
}

// ---------------- host launch ----------------
extern "C" void kernel_launch(void* const* d_in, const int* in_sizes, int n_in,
                              void* d_out, int out_size, void* d_ws, size_t ws_size,
                              hipStream_t stream) {
    (void)in_sizes; (void)n_in; (void)out_size; (void)ws_size;

    const float* x     = (const float*)d_in[0];
    const float* wq    = (const float*)d_in[1];
    const float* wk    = (const float*)d_in[2];
    const float* wv    = (const float*)d_in[3];
    const float* Er    = (const float*)d_in[4];
    const float* fc1_w = (const float*)d_in[5];
    const float* fc1_b = (const float*)d_in[6];
    const float* fc2_w = (const float*)d_in[7];
    const float* fc2_b = (const float*)d_in[8];
    const float* ln1_w = (const float*)d_in[9];
    const float* ln1_b = (const float*)d_in[10];
    const float* ln2_w = (const float*)d_in[11];
    const float* ln2_b = (const float*)d_in[12];

    char* p = (char*)d_ws;
    auto alloc = [&](size_t bytes) -> void* {
        void* r = (void*)p;
        p += (bytes + 255) & ~(size_t)255;
        return r;
    };

    unsigned short* wq_bf  = (unsigned short*)alloc((size_t)DD * DD * 2);
    unsigned short* wk_bf  = (unsigned short*)alloc((size_t)DD * DD * 2);
    unsigned short* wv_bf  = (unsigned short*)alloc((size_t)DD * DD * 2);
    unsigned short* f1w_bf = (unsigned short*)alloc((size_t)D4 * DD * 2);
    unsigned short* f2w_bf = (unsigned short*)alloc((size_t)DD * D4 * 2);
    unsigned short* er_bf  = (unsigned short*)alloc((size_t)HH * TT * HDD * 2);
    unsigned short* xn_bf  = (unsigned short*)alloc((size_t)MM * DD * 2);
    unsigned short* q_bf   = (unsigned short*)alloc((size_t)MM * DD * 2);
    unsigned short* k_bf   = (unsigned short*)alloc((size_t)MM * DD * 2);
    unsigned short* vt_bf  = (unsigned short*)alloc((size_t)MM * DD * 2);
    float*          attb   = (float*)alloc((size_t)MM * DD * 4);
    float*          x2     = (float*)alloc((size_t)MM * DD * 4);
    unsigned short* h_bf   = (unsigned short*)alloc((size_t)MM * DD * 2);
    unsigned short* h2_bf  = (unsigned short*)alloc((size_t)MM * D4 * 2);

    // 1) weights / Er -> bf16
    cvt_bf16_kernel<<<512, 256, 0, stream>>>(wq, wq_bf, DD * DD);
    cvt_bf16_kernel<<<512, 256, 0, stream>>>(wk, wk_bf, DD * DD);
    cvt_bf16_kernel<<<512, 256, 0, stream>>>(wv, wv_bf, DD * DD);
    cvt_bf16_kernel<<<512, 256, 0, stream>>>(fc1_w, f1w_bf, D4 * DD);
    cvt_bf16_kernel<<<512, 256, 0, stream>>>(fc2_w, f2w_bf, DD * D4);
    cvt_bf16_kernel<<<512, 256, 0, stream>>>(Er, er_bf, HH * TT * HDD);

    // 2) LN1 -> bf16 activations
    ln_bf16_kernel<<<MM, 256, 0, stream>>>(x, ln1_w, ln1_b, xn_bf);

    // 3) QKV projections (WMMA), V written transposed
    {
        dim3 grid(DD / 64, MM / 128);
        gemm_nt_bf16<0><<<grid, 256, 0, stream>>>(xn_bf, wq_bf, q_bf, DD, DD, nullptr, nullptr);
        gemm_nt_bf16<0><<<grid, 256, 0, stream>>>(xn_bf, wk_bf, k_bf, DD, DD, nullptr, nullptr);
        gemm_nt_bf16<1><<<grid, 256, 0, stream>>>(xn_bf, wv_bf, vt_bf, DD, DD, nullptr, nullptr);
    }

    // 4) flash attention with relative position skew
    attn_kernel<<<BB * HH * (TT / 16), 32, 0, stream>>>(q_bf, k_bf, vt_bf, er_bf, attb);

    // 5) residual + LN2 -> bf16 (and keep x2 fp32 for final residual)
    resid_ln_bf16_kernel<<<MM, 256, 0, stream>>>(x, attb, ln2_w, ln2_b, x2, h_bf);

    // 6) MLP fc1 (+bias+GELU -> bf16)
    {
        dim3 grid(D4 / 64, MM / 128);
        gemm_nt_bf16<2><<<grid, 256, 0, stream>>>(h_bf, f1w_bf, h2_bf, D4, DD, fc1_b, nullptr);
    }
    // 7) MLP fc2 (+bias+residual -> fp32 d_out)
    {
        dim3 grid(DD / 64, MM / 128);
        gemm_nt_bf16<3><<<grid, 256, 0, stream>>>(h2_bf, f2w_bf, d_out, DD, D4, fc2_b, x2);
    }
}
